// TransformerBlock_38869454029077
// MI455X (gfx1250) — compile-verified, matrix operand order checked by reading, not run on hardware
//
#include <hip/hip_runtime.h>
#include <hip/hip_bf16.h>

// ---------------------------------------------------------------------------
// Problem constants (fixed by the reference)
// ---------------------------------------------------------------------------
#define BB   2
#define TT   2048
#define DD   1024
#define HH   16
#define HD   64          // DD / HH
#define DFF  4096
#define BT   (BB * TT)   // 4096 rows

typedef __attribute__((ext_vector_type(16))) __bf16 v16bf;
typedef __attribute__((ext_vector_type(8)))  float  v8f;

// ---------------------------------------------------------------------------
// Helpers
// ---------------------------------------------------------------------------
static __device__ __forceinline__ unsigned short f2bf(float x) {
    unsigned u = __float_as_uint(x);
    unsigned r = u + 0x7FFFu + ((u >> 16) & 1u);   // round-to-nearest-even
    return (unsigned short)(r >> 16);
}

static __device__ __forceinline__ v8f v8f_zero() {
    v8f z;
#pragma unroll
    for (int i = 0; i < 8; ++i) z[i] = 0.0f;
    return z;
}

static __device__ __forceinline__ v8f wmma_bf16(v16bf a, v16bf b, v8f c) {
    // v_wmma_f32_16x16x32_bf16: D = A(16x32 bf16) * B(32x16 bf16) + C(f32)
    return __builtin_amdgcn_wmma_f32_16x16x32_bf16(
        /*neg_a=*/false, a, /*neg_b=*/false, b,
        /*c_mod=*/(short)0, c, /*reuse_a=*/false, /*reuse_b=*/false);
}

union Frag16 {
    uint4 u[2];
    v16bf v;
};

// Load one 16-bit WMMA fragment when the two 8-element K-runs (e0..7 at +0,
// e8..15 at +16 elements) are contiguous in memory. `base` points at the
// lane's row/col start + klo. Works for global and LDS pointers.
static __device__ __forceinline__ v16bf load_frag(const unsigned short* base) {
    Frag16 f;
    f.u[0] = *reinterpret_cast<const uint4*>(base);
    f.u[1] = *reinterpret_cast<const uint4*>(base + 16);
    return f.v;
}

static __device__ __forceinline__ float gelu_tanh(float x) {
    float x3 = x * x * x;
    return 0.5f * x * (1.0f + tanhf(0.7978845608028654f * (x + 0.044715f * x3)));
}

// ---------------------------------------------------------------------------
// Kernel: transpose-convert weights  W[K,N] f32  ->  Wt[N,K] bf16
// ---------------------------------------------------------------------------
__global__ __launch_bounds__(256) void wtrans_bf16_kernel(
    const float* __restrict__ W, unsigned short* __restrict__ Wt,
    int K, int N) {
    size_t idx = (size_t)blockIdx.x * 256 + threadIdx.x;
    size_t total = (size_t)K * N;
    if (idx >= total) return;
    int kk = (int)(idx / N);
    int n  = (int)(idx % N);
    Wt[(size_t)n * K + kk] = f2bf(W[idx]);
}

// ---------------------------------------------------------------------------
// Kernel: repack bf16 V [B*T, D] -> vt[b][h][d][t]  (t contiguous) so the
// attention PV B-fragments become two contiguous b128 loads per fragment.
// ---------------------------------------------------------------------------
__global__ __launch_bounds__(256) void vtrans_kernel(
    const unsigned short* __restrict__ vin, unsigned short* __restrict__ vout) {
    size_t idx = (size_t)blockIdx.x * 256 + threadIdx.x;   // over BT*DD
    int row = (int)(idx / DD);
    int col = (int)(idx % DD);
    int b = row / TT, t = row % TT;
    int h = col / HD, d = col % HD;
    vout[(size_t)((b * HH + h) * HD + d) * TT + t] = vin[idx];
}

// ---------------------------------------------------------------------------
// Kernel: LayerNorm rows of f32 [rows, ncols] -> bf16
// ---------------------------------------------------------------------------
__global__ __launch_bounds__(256) void ln_bf16_kernel(
    const float* __restrict__ x, const float* __restrict__ g,
    const float* __restrict__ bta, unsigned short* __restrict__ out,
    int ncols) {
    __shared__ float rs[256];
    __shared__ float rss[256];
    const int row = blockIdx.x;
    const int tid = threadIdx.x;
    const float* xr = x + (size_t)row * ncols;

    float s = 0.f, ss = 0.f;
    for (int c = tid; c < ncols; c += 256) {
        float vv = xr[c];
        s += vv;
        ss += vv * vv;
    }
    rs[tid] = s; rss[tid] = ss;
    __syncthreads();
#pragma unroll
    for (int off = 128; off > 0; off >>= 1) {
        if (tid < off) { rs[tid] += rs[tid + off]; rss[tid] += rss[tid + off]; }
        __syncthreads();
    }
    const float mu   = rs[0] / (float)ncols;
    const float var  = rss[0] / (float)ncols - mu * mu;
    const float rstd = rsqrtf(var + 1e-5f);

    unsigned short* orow = out + (size_t)row * ncols;
    for (int c = tid; c < ncols; c += 256) {
        orow[c] = f2bf((xr[c] - mu) * rstd * g[c] + bta[c]);
    }
}

// ---------------------------------------------------------------------------
// Kernel: GEMM  C[M,N] = A[M,K](bf16,row-major) * Bt[N,K](bf16) + bias
// optional GELU; optional f32 and/or bf16 outputs.
// Block: 256 threads = 8 waves (2x4). Wave tile 32(M) x 64(N): 2x4 WMMA
// accumulators, 8 WMMAs per 32-deep K step from 6 fragments (12 b128 loads).
// Block tile 64(M) x 256(N).
// ---------------------------------------------------------------------------
__global__ __launch_bounds__(256) void gemm_bf16_wmma_kernel(
    const unsigned short* __restrict__ A,
    const unsigned short* __restrict__ Bt,
    const float* __restrict__ bias,
    float* __restrict__ outF,
    unsigned short* __restrict__ outB,
    int M, int N, int K, int act) {
    const int lane = threadIdx.x & 31;
    const int w    = threadIdx.x >> 5;
    const int wm   = w >> 2;        // 0..1
    const int wn   = w & 3;         // 0..3
    const int mBase = blockIdx.y * 64 + wm * 32;
    const int nBase = blockIdx.x * 256 + wn * 64;

    const int l15  = lane & 15;
    const int kHi  = (lane & 16) ? 8 : 0;

    const unsigned short* aRow0 = A + (size_t)(mBase + l15)      * K;
    const unsigned short* aRow1 = A + (size_t)(mBase + 16 + l15) * K;
    const unsigned short* bRow[4];
#pragma unroll
    for (int ni = 0; ni < 4; ++ni)
        bRow[ni] = Bt + (size_t)(nBase + ni * 16 + l15) * K;

    v8f acc[2][4];
#pragma unroll
    for (int mi = 0; mi < 2; ++mi)
#pragma unroll
        for (int ni = 0; ni < 4; ++ni) acc[mi][ni] = v8f_zero();

    for (int kb = 0; kb < K; kb += 32) {
        const int klo = kb + kHi;
        if (kb + 64 < K) {
            __builtin_prefetch(aRow0 + klo + 64, 0, 3);
            __builtin_prefetch(bRow[0] + klo + 64, 0, 3);
        }
        const v16bf a0 = load_frag(aRow0 + klo);
        const v16bf a1 = load_frag(aRow1 + klo);
        v16bf bf[4];
#pragma unroll
        for (int ni = 0; ni < 4; ++ni) bf[ni] = load_frag(bRow[ni] + klo);
#pragma unroll
        for (int ni = 0; ni < 4; ++ni) {
            acc[0][ni] = wmma_bf16(a0, bf[ni], acc[0][ni]);
            acc[1][ni] = wmma_bf16(a1, bf[ni], acc[1][ni]);
        }
    }

#pragma unroll
    for (int ni = 0; ni < 4; ++ni) {
        const float bc = bias[nBase + ni * 16 + l15];
        const int col = nBase + ni * 16 + l15;
#pragma unroll
        for (int mi = 0; mi < 2; ++mi) {
#pragma unroll
            for (int r = 0; r < 8; ++r) {
                const int row = mBase + mi * 16 + r + ((lane & 16) ? 8 : 0);
                float vv = acc[mi][ni][r] + bc;
                if (act) vv = gelu_tanh(vv);
                const size_t o = (size_t)row * N + col;
                if (outF) outF[o] = vv;
                if (outB) outB[o] = f2bf(vv);
            }
        }
    }
}

// ---------------------------------------------------------------------------
// Kernel: causal flash attention.
//   Q,K: bf16 [B*T, D] (head-interleaved), V: bf16 vt[b][h][d][t],
//   output: bf16 [B*T, D].
// One wave handles one 16-row Q tile of one head; online softmax over
// 32-key steps; P re-layout through a per-wave LDS slab.
// Grid: B*H*(T/16)/8 blocks of 256 threads.
// ---------------------------------------------------------------------------
__global__ __launch_bounds__(256) void attn_wmma_kernel(
    const unsigned short* __restrict__ q,
    const unsigned short* __restrict__ k,
    const unsigned short* __restrict__ vt,
    unsigned short* __restrict__ o) {
    __shared__ __align__(16) unsigned short plds[8][16 * 32];  // 1KB / wave

    const int lane = threadIdx.x & 31;
    const int w    = threadIdx.x >> 5;
    const int gw   = blockIdx.x * 8 + w;
    const int qtiles = TT / 16;               // 128
    const int bh = gw / qtiles;
    const int i  = gw % qtiles;               // q tile index
    const int b  = bh / HH;
    const int h  = bh % HH;

    const size_t rowbase = (size_t)b * TT;
    const int    ch      = h * HD;            // head column offset in D
    const int    l15     = lane & 15;
    const int    kHi     = (lane & 16) ? 8 : 0;
    const int    rHi     = (lane & 16) ? 8 : 0;

    // Q A-fragments for the full hd=64 (two 32-deep K chunks)
    const unsigned short* qrow = q + (rowbase + i * 16 + l15) * DD + ch;
    const v16bf qa0 = load_frag(qrow + 0  + kHi);
    const v16bf qa1 = load_frag(qrow + 32 + kHi);

    // V (transposed) row pointers: vt[bh][d][t], lane owns column d
    const unsigned short* vRow[4];
#pragma unroll
    for (int nt = 0; nt < 4; ++nt)
        vRow[nt] = vt + (size_t)(bh * HD + nt * 16 + l15) * TT;

    float m[8], l[8];
    v8f o4[4];
#pragma unroll
    for (int r = 0; r < 8; ++r) { m[r] = -1e30f; l[r] = 0.0f; }
#pragma unroll
    for (int nt = 0; nt < 4; ++nt) o4[nt] = v8f_zero();

    unsigned short* myp = plds[w];
    const int lastkey = i * 16 + 15;

    for (int kb0 = 0; kb0 <= lastkey; kb0 += 32) {
        // ---- S = (Q K^T) * 1/sqrt(hd), two 16x16 key tiles -----------------
        v8f s[2];
#pragma unroll
        for (int st = 0; st < 2; ++st) {
            const int keyc = kb0 + st * 16 + l15;
            const unsigned short* krow = k + (rowbase + keyc) * DD + ch;
            v8f c = v8f_zero();
            c = wmma_bf16(qa0, load_frag(krow + 0  + kHi), c);
            c = wmma_bf16(qa1, load_frag(krow + 32 + kHi), c);
            s[st] = c;
        }

        // ---- causal mask + online softmax (rows live per reg, cols per lane)
#pragma unroll
        for (int r = 0; r < 8; ++r) {
            const int qt = i * 16 + r + rHi;
            float s0 = s[0][r] * 0.125f;                 // 1/sqrt(64)
            float s1 = s[1][r] * 0.125f;
            const int k0 = kb0 + l15;
            const int k1 = k0 + 16;
            if (k0 > qt) s0 = -1e30f;
            if (k1 > qt) s1 = -1e30f;

            float vm = fmaxf(s0, s1);
#pragma unroll
            for (int off = 8; off > 0; off >>= 1)
                vm = fmaxf(vm, __shfl_xor(vm, off, 32));
            const float nm = fmaxf(m[r], vm);
            const float p0 = __expf(s0 - nm);
            const float p1 = __expf(s1 - nm);
            const float alpha = __expf(m[r] - nm);
            m[r] = nm;
            float rsum = p0 + p1;
#pragma unroll
            for (int off = 8; off > 0; off >>= 1)
                rsum += __shfl_xor(rsum, off, 32);
            l[r] = l[r] * alpha + rsum;
#pragma unroll
            for (int nt = 0; nt < 4; ++nt) o4[nt][r] *= alpha;

            const int prow = r + rHi;
            myp[prow * 32 + l15]      = f2bf(p0);
            myp[prow * 32 + 16 + l15] = f2bf(p1);
        }

        // ---- P (16x32) from LDS in A-fragment layout -----------------------
        const v16bf pa = load_frag(myp + l15 * 32 + kHi);

        // ---- O += P @ V_block (32 keys x 64 hd, four 16-col tiles) ---------
        // vt layout makes each B-fragment two contiguous b128 loads.
#pragma unroll
        for (int nt = 0; nt < 4; ++nt) {
            const v16bf vfr = load_frag(vRow[nt] + kb0 + kHi);
            o4[nt] = wmma_bf16(pa, vfr, o4[nt]);
        }
    }

    // ---- finalize: O /= l, store bf16 -------------------------------------
#pragma unroll
    for (int r = 0; r < 8; ++r) {
        const float inv = 1.0f / l[r];
        const int t = i * 16 + r + rHi;
#pragma unroll
        for (int nt = 0; nt < 4; ++nt) {
            o[(rowbase + t) * DD + ch + nt * 16 + l15] = f2bf(o4[nt][r] * inv);
        }
    }
}

// ---------------------------------------------------------------------------
// Host launcher
// ---------------------------------------------------------------------------
extern "C" void kernel_launch(void* const* d_in, const int* in_sizes, int n_in,
                              void* d_out, int out_size, void* d_ws, size_t ws_size,
                              hipStream_t stream) {
    (void)in_sizes; (void)n_in; (void)out_size; (void)ws_size;

    const float* x     = (const float*)d_in[0];
    const float* ln1_g = (const float*)d_in[1];
    const float* ln1_b = (const float*)d_in[2];
    const float* Wq    = (const float*)d_in[3];
    const float* bq    = (const float*)d_in[4];
    const float* Wk    = (const float*)d_in[5];
    const float* bk    = (const float*)d_in[6];
    const float* Wv    = (const float*)d_in[7];
    const float* bv    = (const float*)d_in[8];
    const float* Wc    = (const float*)d_in[9];
    const float* bc    = (const float*)d_in[10];
    const float* ln2_g = (const float*)d_in[11];
    const float* ln2_b = (const float*)d_in[12];
    const float* Wfc   = (const float*)d_in[13];
    const float* bfc   = (const float*)d_in[14];
    const float* Wpr   = (const float*)d_in[15];
    const float* bpr   = (const float*)d_in[16];

    // ---- workspace carve-out ----------------------------------------------
    char* p = (char*)d_ws;
    auto alloc = [&](size_t bytes) -> void* {
        void* r = (void*)p;
        p += (bytes + 255) & ~(size_t)255;
        return r;
    };
    unsigned short* wq_t  = (unsigned short*)alloc((size_t)DD * DD * 2);
    unsigned short* wk_t  = (unsigned short*)alloc((size_t)DD * DD * 2);
    unsigned short* wv_t  = (unsigned short*)alloc((size_t)DD * DD * 2);
    unsigned short* wc_t  = (unsigned short*)alloc((size_t)DD * DD * 2);
    unsigned short* wfc_t = (unsigned short*)alloc((size_t)DD * DFF * 2);
    unsigned short* wpr_t = (unsigned short*)alloc((size_t)DFF * DD * 2);
    unsigned short* h1    = (unsigned short*)alloc((size_t)BT * DD * 2);
    unsigned short* qb    = (unsigned short*)alloc((size_t)BT * DD * 2);
    unsigned short* kb    = (unsigned short*)alloc((size_t)BT * DD * 2);
    unsigned short* vb    = (unsigned short*)alloc((size_t)BT * DD * 2);
    unsigned short* vtb   = (unsigned short*)alloc((size_t)BT * DD * 2);
    unsigned short* ab    = (unsigned short*)alloc((size_t)BT * DD * 2);
    float*          yproj = (float*)         alloc((size_t)BT * DD * 4);
    unsigned short* h2    = (unsigned short*)alloc((size_t)BT * DD * 2);
    unsigned short* ff    = (unsigned short*)alloc((size_t)BT * DFF * 2);

    // ---- 1) weight transpose+convert to bf16 ------------------------------
    {
        int blocks_dd  = (DD * DD) / 256;
        int blocks_dff = (DD * DFF) / 256;
        wtrans_bf16_kernel<<<blocks_dd,  256, 0, stream>>>(Wq,  wq_t,  DD,  DD);
        wtrans_bf16_kernel<<<blocks_dd,  256, 0, stream>>>(Wk,  wk_t,  DD,  DD);
        wtrans_bf16_kernel<<<blocks_dd,  256, 0, stream>>>(Wv,  wv_t,  DD,  DD);
        wtrans_bf16_kernel<<<blocks_dd,  256, 0, stream>>>(Wc,  wc_t,  DD,  DD);
        wtrans_bf16_kernel<<<blocks_dff, 256, 0, stream>>>(Wfc, wfc_t, DD,  DFF);
        wtrans_bf16_kernel<<<blocks_dff, 256, 0, stream>>>(Wpr, wpr_t, DFF, DD);
    }

    // ---- 2) LN1 -----------------------------------------------------------
    ln_bf16_kernel<<<BT, 256, 0, stream>>>(x, ln1_g, ln1_b, h1, DD);

    // ---- 3) QKV projections -----------------------------------------------
    {
        dim3 grid(DD / 256, BT / 64);
        gemm_bf16_wmma_kernel<<<grid, 256, 0, stream>>>(h1, wq_t, bq, nullptr, qb, BT, DD, DD, 0);
        gemm_bf16_wmma_kernel<<<grid, 256, 0, stream>>>(h1, wk_t, bk, nullptr, kb, BT, DD, DD, 0);
        gemm_bf16_wmma_kernel<<<grid, 256, 0, stream>>>(h1, wv_t, bv, nullptr, vb, BT, DD, DD, 0);
    }

    // ---- 4) repack V head-major transposed for contiguous PV fragments ----
    vtrans_kernel<<<(BT * DD) / 256, 256, 0, stream>>>(vb, vtb);

    // ---- 5) causal flash attention ----------------------------------------
    {
        int blocks = (BB * HH * (TT / 16)) / 8;   // 512
        attn_wmma_kernel<<<blocks, 256, 0, stream>>>(qb, kb, vtb, ab);
    }

    // ---- 6) output projection (f32 out for LN2) ---------------------------
    {
        dim3 grid(DD / 256, BT / 64);
        gemm_bf16_wmma_kernel<<<grid, 256, 0, stream>>>(ab, wc_t, bc, yproj, nullptr, BT, DD, DD, 0);
    }

    // ---- 7) LN2 -----------------------------------------------------------
    ln_bf16_kernel<<<BT, 256, 0, stream>>>(yproj, ln2_g, ln2_b, h2, DD);

    // ---- 8) MLP: GELU(h2 @ Wfc + bfc) -------------------------------------
    {
        dim3 grid(DFF / 256, BT / 64);
        gemm_bf16_wmma_kernel<<<grid, 256, 0, stream>>>(h2, wfc_t, bfc, nullptr, ff, BT, DFF, DD, 1);
    }

    // ---- 9) MLP projection -> d_out ---------------------------------------
    {
        dim3 grid(DD / 256, BT / 64);
        gemm_bf16_wmma_kernel<<<grid, 256, 0, stream>>>(ff, wpr_t, bpr, (float*)d_out, nullptr, BT, DD, DFF, 0);
    }
}
